// SimilarityGuidedSampling_38749194944693
// MI455X (gfx1250) — matrix operand order repaired and checked by reading, not compiled, not run on hardware
//
#include <hip/hip_runtime.h>
#include <hip/hip_bf16.h>
#include <math.h>

// ---------------------------------------------------------------------------
// Problem constants (from reference): B=8, C=512, T=32, H=W=14, hidden=1024,
// E=32, NUM_BINS=8.  M = B*T = 256 "rows" for the two GEMMs.
// ---------------------------------------------------------------------------
#define BB   8
#define CC   512
#define TT   32
#define HWD  196          // 14*14
#define HWD4 49           // 196/4 float4 quadrants (row stride 784B = 49*16B, aligned)
#define HID  1024
#define EE   32
#define MM   256          // B*T
#define NBIN 8

typedef __attribute__((ext_vector_type(16))) __bf16 v16bf;
typedef __attribute__((ext_vector_type(8)))  __bf16 v8bf;
typedef __attribute__((ext_vector_type(8)))  float  v8f;
typedef __attribute__((ext_vector_type(4)))  float  f32x4;

// ---------------------------------------------------------------------------
// K0: convert w1 (1024x512) and w2 (32x1024) fp32 -> bf16 (row-major kept:
// row index = N (output feature), inner index = K -> exactly the per-lane
// contiguous K-run the WMMA B-operand layout wants).
// ---------------------------------------------------------------------------
__global__ void cvt_weights_kernel(const float* __restrict__ w1,
                                   const float* __restrict__ w2,
                                   __bf16* __restrict__ w1b,
                                   __bf16* __restrict__ w2b) {
    int i = blockIdx.x * 256 + threadIdx.x;
    if (i < HID * CC)  w1b[i] = (__bf16)w1[i];
    if (i < EE * HID)  w2b[i] = (__bf16)w2[i];
}

// ---------------------------------------------------------------------------
// K1: spatial avg+max pool.  One wave32 per (b,c,t): lanes stride over the
// 49 aligned float4 quadrants of the 196-element spatial plane (coalesced
// global_load_b128), then a 5-step __shfl_xor wave reduction.
// Result stored bf16 as A-matrix [m=b*T+t][k=c].  Loads use default RT
// temporal hint so x stays resident in the 192MB L2 for the scatter kernel.
// ---------------------------------------------------------------------------
__global__ void pool_kernel(const float* __restrict__ x,
                            __bf16* __restrict__ pooled) {
    int wave = (blockIdx.x * blockDim.x + threadIdx.x) >> 5;   // 131072 waves
    int lane = threadIdx.x & 31;
    int t = wave & 31;
    int c = (wave >> 5) & 511;
    int b = wave >> 14;
    const f32x4* p4 =
        (const f32x4*)(x + (size_t)((b * CC + c) * TT + t) * HWD);
    float s  = 0.0f;
    float mx = -3.402823466e+38f;
    for (int i = lane; i < HWD4; i += 32) {   // 2 iterations, 2nd predicated
        f32x4 v = p4[i];
        s += v[0] + v[1] + v[2] + v[3];
        mx = fmaxf(mx, fmaxf(fmaxf(v[0], v[1]), fmaxf(v[2], v[3])));
    }
    for (int off = 16; off; off >>= 1) {
        s  += __shfl_xor(s, off, 32);
        mx  = fmaxf(mx, __shfl_xor(mx, off, 32));
    }
    if (lane == 0) {
        int m = b * TT + t;
        pooled[(size_t)m * CC + c] = (__bf16)(s * (1.0f / (float)HWD) + mx);
    }
}

// ---------------------------------------------------------------------------
// K2: GEMM1  h[m][n] = hswish( sum_k pooled[m][k] * w1[n][k] + b1[n] )
// M=256, N=1024, K=512 via v_wmma_f32_16x16x32_bf16.
// One wave per 16x16 output tile (16 m-tiles x 64 n-tiles = 1024 waves),
// 16 k-steps of K=32 each.
// A operand layout (16-bit 16x32, ISA 7.12.2): lane = half*16 + row,
//   elems 0..7  = A[row][k0 + half*8 .. +7]
//   elems 8..15 = A[row][k0 + 16 + half*8 .. +7]
// B operand layout (32x16): lane = half*16 + col,
//   elems 0..7  = B[k0 + half*8 .. +7][col]
//   elems 8..15 = B[k0 + 16 + half*8 .. +7][col]  (w1 held as [n][k])
// C/D layout: VGPR r holds row (r + half*8), col = lane&15.
// ---------------------------------------------------------------------------
__global__ void gemm1_hswish_kernel(const __bf16* __restrict__ A,
                                    const __bf16* __restrict__ Bw,
                                    const float* __restrict__ b1,
                                    __bf16* __restrict__ Hout) {
    int wave = (blockIdx.x * blockDim.x + threadIdx.x) >> 5;   // 1024 waves
    int lane = threadIdx.x & 31;
    int nt = wave & 63;
    int mt = wave >> 6;
    int m0 = mt << 4, n0 = nt << 4;
    int half   = lane >> 4;
    int lane16 = lane & 15;

    v8f acc = {0.f, 0.f, 0.f, 0.f, 0.f, 0.f, 0.f, 0.f};
    const __bf16* arow = A  + (size_t)(m0 + lane16) * CC;
    const __bf16* brow = Bw + (size_t)(n0 + lane16) * CC;

    for (int k0 = 0; k0 < CC; k0 += 32) {
        v8bf a_lo = *(const v8bf*)(arow + k0 + half * 8);
        v8bf a_hi = *(const v8bf*)(arow + k0 + 16 + half * 8);
        v8bf b_lo = *(const v8bf*)(brow + k0 + half * 8);
        v8bf b_hi = *(const v8bf*)(brow + k0 + 16 + half * 8);
        v16bf av = __builtin_shufflevector(a_lo, a_hi,
            0,1,2,3,4,5,6,7,8,9,10,11,12,13,14,15);
        v16bf bv = __builtin_shufflevector(b_lo, b_hi,
            0,1,2,3,4,5,6,7,8,9,10,11,12,13,14,15);
        acc = __builtin_amdgcn_wmma_f32_16x16x32_bf16(
            false, av, false, bv, (short)0, acc, false, false);
    }

    int n = n0 + lane16;
    float bias = b1[n];
    for (int r = 0; r < 8; ++r) {
        int m = m0 + r + half * 8;
        float h  = acc[r] + bias;
        float hs = h * fminf(fmaxf(h + 3.0f, 0.0f), 6.0f) * (1.0f / 6.0f);
        Hout[(size_t)m * HID + n] = (__bf16)hs;
    }
}

// ---------------------------------------------------------------------------
// K3: GEMM2  e[m][n] = sum_k h[m][k] * w2[n][k] + b2[n]
// M=256, N=32, K=1024 -> 16 m-tiles x 2 n-tiles = 32 waves, 32 k-steps.
// ---------------------------------------------------------------------------
__global__ void gemm2_kernel(const __bf16* __restrict__ Hm,
                             const __bf16* __restrict__ Bw,
                             const float* __restrict__ b2,
                             float* __restrict__ embds) {
    int wave = (blockIdx.x * blockDim.x + threadIdx.x) >> 5;   // 32 waves
    int lane = threadIdx.x & 31;
    int nt = wave & 1;
    int mt = wave >> 1;
    int m0 = mt << 4, n0 = nt << 4;
    int half   = lane >> 4;
    int lane16 = lane & 15;

    v8f acc = {0.f, 0.f, 0.f, 0.f, 0.f, 0.f, 0.f, 0.f};
    const __bf16* arow = Hm + (size_t)(m0 + lane16) * HID;
    const __bf16* brow = Bw + (size_t)(n0 + lane16) * HID;

    for (int k0 = 0; k0 < HID; k0 += 32) {
        v8bf a_lo = *(const v8bf*)(arow + k0 + half * 8);
        v8bf a_hi = *(const v8bf*)(arow + k0 + 16 + half * 8);
        v8bf b_lo = *(const v8bf*)(brow + k0 + half * 8);
        v8bf b_hi = *(const v8bf*)(brow + k0 + 16 + half * 8);
        v16bf av = __builtin_shufflevector(a_lo, a_hi,
            0,1,2,3,4,5,6,7,8,9,10,11,12,13,14,15);
        v16bf bv = __builtin_shufflevector(b_lo, b_hi,
            0,1,2,3,4,5,6,7,8,9,10,11,12,13,14,15);
        acc = __builtin_amdgcn_wmma_f32_16x16x32_bf16(
            false, av, false, bv, (short)0, acc, false, false);
    }

    int n = n0 + lane16;
    float bias = b2[n];
    for (int r = 0; r < 8; ++r) {
        int m = m0 + r + half * 8;
        embds[(size_t)m * EE + n] = acc[r] + bias;
    }
}

// ---------------------------------------------------------------------------
// K4: per-batch group construction.  One wave32 per batch, lane = t.
// L2-normalize over E, neighbour cosine sims via LDS, then serial top-7
// smallest (stable on ties) on lane 0.  cumsum(breaks) is non-decreasing so
// bins are contiguous [start,end) ranges: emit bin_start[9] + bin_scale[8].
// ---------------------------------------------------------------------------
__global__ void group_kernel(const float* __restrict__ embds,
                             int* __restrict__ bin_start,
                             float* __restrict__ bin_scale) {
    int b = blockIdx.x;
    int t = threadIdx.x;          // 0..31
    __shared__ float sn[TT][EE + 1];
    __shared__ float ssim[TT];

    const float* e = embds + (size_t)(b * TT + t) * EE;
    float v[EE];
    float ss = 0.0f;
    for (int i = 0; i < EE; ++i) { v[i] = e[i]; ss += v[i] * v[i]; }
    float inv = 1.0f / fmaxf(sqrtf(ss), 1e-12f);
    for (int i = 0; i < EE; ++i) sn[t][i] = v[i] * inv;
    __syncthreads();

    float sim = 0.0f;
    if (t >= 1)
        for (int i = 0; i < EE; ++i) sim += sn[t][i] * sn[t - 1][i];
    ssim[t] = sim;
    __syncthreads();

    if (t == 0) {
        bool used[TT - 1];
        int  brk[TT - 1];
        for (int j = 0; j < TT - 1; ++j) { used[j] = false; brk[j] = 0; }
        for (int it = 0; it < NBIN - 1; ++it) {
            int am = 0; float best = 3.402823466e+38f;
            for (int j = 0; j < TT - 1; ++j)
                if (!used[j] && ssim[j + 1] < best) { best = ssim[j + 1]; am = j; }
            used[am] = true; brk[am] = 1;
        }
        int start[NBIN + 1];
        start[0] = 0;
        int g = 0;
        for (int tt = 1; tt < TT; ++tt)
            if (brk[tt - 1]) start[++g] = tt;
        for (int n = g + 1; n <= NBIN; ++n) start[n] = TT;
        for (int n = 0; n < NBIN + 1; ++n) bin_start[b * (NBIN + 1) + n] = start[n];
        for (int n = 0; n < NBIN; ++n) {
            int sz = start[n + 1] - start[n];
            bin_scale[b * NBIN + n] = (sz > 0) ? (1.0f / (float)sz) : 1.0f;
        }
    }
}

// ---------------------------------------------------------------------------
// K5: grouped temporal mean scatter, float4-vectorized over hw.
// out[b][c][n][hw] = scale[b][n] * sum_{t in [start[n],start[n+1])} x[...]
// One thread per (b,c,hw-quadrant): 32 b128 loads (x should be L2-resident
// from the pool kernel) + 8 NONTEMPORAL b128 stores (output is write-once,
// never re-read -> keep it from evicting x out of L2).
// Block = 256 threads; C*HWD4 = 25088 is divisible by 256 so a block never
// straddles a batch boundary.
// ---------------------------------------------------------------------------
__global__ void scatter_kernel(const float* __restrict__ x,
                               const int* __restrict__ bin_start,
                               const float* __restrict__ bin_scale,
                               float* __restrict__ out) {
    __shared__ int   st[NBIN + 1];
    __shared__ float sc[NBIN];
    long flat = (long)blockIdx.x * 256 + threadIdx.x;   // < B*C*49 = 200704
    int b = (int)(flat / (CC * HWD4));
    int r = (int)(flat % (CC * HWD4));
    int c = r / HWD4;
    int q = r % HWD4;                                   // float4 quadrant

    if (threadIdx.x < NBIN + 1) st[threadIdx.x] = bin_start[b * (NBIN + 1) + threadIdx.x];
    if (threadIdx.x < NBIN)     sc[threadIdx.x] = bin_scale[b * NBIN + threadIdx.x];
    __syncthreads();

    const f32x4* xp4 =
        (const f32x4*)(x + (size_t)((b * CC + c) * TT) * HWD) + q;
    f32x4* op4 =
        (f32x4*)(out + (size_t)((b * CC + c) * NBIN) * HWD) + q;

    for (int n = 0; n < NBIN; ++n) {
        f32x4 acc = {0.f, 0.f, 0.f, 0.f};
        int e0 = st[n], e1 = st[n + 1];
        for (int t = e0; t < e1; ++t) {
            f32x4 v = xp4[(size_t)t * HWD4];
            acc[0] += v[0]; acc[1] += v[1]; acc[2] += v[2]; acc[3] += v[3];
        }
        float s = sc[n];
        acc[0] *= s; acc[1] *= s; acc[2] *= s; acc[3] *= s;
        __builtin_nontemporal_store(acc, &op4[(size_t)n * HWD4]);
    }
}

// ---------------------------------------------------------------------------
// Launcher.  Inputs: x, w1, b1, w2, b2 (all fp32).  Output: (B,C,8,H,W) fp32.
// Workspace layout (bytes, 256-aligned):
//   w1b   bf16 1024x512  : 1,048,576
//   w2b   bf16   32x1024 :    65,536
//   pooled bf16 256x512  :   262,144
//   h     bf16 256x1024  :   524,288
//   embds f32  256x32    :    32,768
//   bin_start int 8x9    :       512 (padded)
//   bin_scale f32 8x8    :       256
// ---------------------------------------------------------------------------
extern "C" void kernel_launch(void* const* d_in, const int* in_sizes, int n_in,
                              void* d_out, int out_size, void* d_ws, size_t ws_size,
                              hipStream_t stream) {
    const float* x  = (const float*)d_in[0];
    const float* w1 = (const float*)d_in[1];
    const float* b1 = (const float*)d_in[2];
    const float* w2 = (const float*)d_in[3];
    const float* b2 = (const float*)d_in[4];
    float* out = (float*)d_out;

    char* ws = (char*)d_ws;
    size_t off = 0;
    __bf16* w1b    = (__bf16*)(ws + off); off += (size_t)HID * CC * 2;   // 1 MB
    __bf16* w2b    = (__bf16*)(ws + off); off += (size_t)EE * HID * 2;   // 64 KB
    __bf16* pooled = (__bf16*)(ws + off); off += (size_t)MM * CC * 2;    // 256 KB
    __bf16* hbuf   = (__bf16*)(ws + off); off += (size_t)MM * HID * 2;   // 512 KB
    float*  embds  = (float*)(ws + off);  off += (size_t)MM * EE * 4;    // 32 KB
    int*    bstart = (int*)(ws + off);    off += 512;
    float*  bscale = (float*)(ws + off);  off += 256;

    // K0: weight conversion (524288 elems / 256 = 2048 blocks)
    cvt_weights_kernel<<<2048, 256, 0, stream>>>(w1, w2, w1b, w2b);

    // K1: pool. 131072 waves * 32 lanes / 256 threads = 16384 blocks
    pool_kernel<<<16384, 256, 0, stream>>>(x, pooled);

    // K2: GEMM1 + hswish. 1024 waves -> 128 blocks of 8 waves
    gemm1_hswish_kernel<<<128, 256, 0, stream>>>(pooled, w1b, b1, hbuf);

    // K3: GEMM2 + bias. 32 waves -> 4 blocks of 8 waves
    gemm2_kernel<<<4, 256, 0, stream>>>(hbuf, w2b, b2, embds);

    // K4: per-batch grouping. 8 blocks of one wave32
    group_kernel<<<BB, 32, 0, stream>>>(embds, bstart, bscale);

    // K5: grouped scatter-sum (float4). 200704 threads / 256 = 784 blocks
    scatter_kernel<<<784, 256, 0, stream>>>(x, bstart, bscale, out);
}
